// MultiHeadAttention_32186484917056
// MI455X (gfx1250) — matrix operand order repaired and checked by reading, not compiled, not run on hardware
//
#include <hip/hip_runtime.h>
#include <math.h>

// ---------------------------------------------------------------------------
// MHA for MI455X (gfx1250): bf16 WMMA everywhere, flash-attention softmax,
// async global->LDS double-buffered GEMM main loops.
// ---------------------------------------------------------------------------

typedef __attribute__((ext_vector_type(16))) __bf16 v16bf;
typedef __attribute__((ext_vector_type(8)))  __bf16 v8bf;
typedef __attribute__((ext_vector_type(8)))  float  v8f;

#define B_   16
#define S_   1024
#define D_   768
#define H_   12
#define HD_  64
#define MTOT (B_ * S_)   // 16384
#define LDT  72          // padded LDS stride (bf16 elements), keeps 16B align

#define WMMA_BF16(a, b, c) \
  __builtin_amdgcn_wmma_f32_16x16x32_bf16(false, (a), false, (b), (short)0, (c), false, false)

// Async 16B/lane global->LDS copy (GLOBAL_LOAD_ASYNC_TO_LDS_B128, ASYNCcnt).
// lds_off is the low 32 bits of the generic shared address (== LDS offset).
__device__ __forceinline__ void async_copy_b128(unsigned lds_off, const __bf16* src) {
  asm volatile("global_load_async_to_lds_b128 %0, %1, off"
               :: "v"(lds_off), "v"(src)
               : "memory");
}
#define WAIT_ASYNC(n) asm volatile("s_wait_asynccnt " #n ::: "memory")

__device__ __forceinline__ unsigned lds_off32(const void* p) {
  return (unsigned)(size_t)p;  // LDS aperture: addr[31:0] is the LDS offset
}

// A fragment: 16x32 bf16, lane L(0-15)=row, K groups per ISA 7.12.2
__device__ __forceinline__ v16bf frag_a(const __bf16* base, int ld) {
  const int lane = threadIdx.x & 31;
  const __bf16* p = base + (lane & 15) * ld + ((lane >> 4) & 1) * 8;
  union { v16bf v; v8bf h[2]; } u;
  u.h[0] = *(const v8bf*)(p);        // K = hi*8 + 0..7
  u.h[1] = *(const v8bf*)(p + 16);   // K = hi*8 + 16..23
  return u.v;
}

// B fragment: 32x16 bf16 stored [N][K] in LDS; K = hi*8 + 0..7, hi*8 + 16..23
__device__ __forceinline__ v16bf frag_b(const __bf16* base, int ld) {
  const int lane = threadIdx.x & 31;
  const __bf16* p = base + (lane & 15) * ld + ((lane >> 4) & 1) * 8;
  union { v16bf v; v8bf h[2]; } u;
  u.h[0] = *(const v8bf*)(p);
  u.h[1] = *(const v8bf*)(p + 16);
  return u.v;
}

// ---------------------------------------------------------------------------
__global__ __launch_bounds__(256) void cvt_bf16(const float* __restrict__ src,
                                                __bf16* __restrict__ dst, int n) {
  int i = blockIdx.x * 256 + threadIdx.x;
  if (i < n) dst[i] = (__bf16)src[i];
}

// ---------------------------------------------------------------------------
// Double-buffered async WMMA GEMM core: 64x64 C tile, K = 768 in 64-steps.
// 8 waves arranged 4(M) x 2(N); each wave two 16x16 f32 accumulators.
// ---------------------------------------------------------------------------
__device__ __forceinline__ void gemm64_async(
    const __bf16* __restrict__ gA, const __bf16* __restrict__ gB,
    __bf16 (*As)[64 * LDT], __bf16 (*Bs)[64 * LDT], v8f& acc0, v8f& acc1) {
  const int tid = threadIdx.x;
  const int wave = tid >> 5, wm = wave >> 1, wn = wave & 1;
  const int lrow = tid >> 3, lcol = (tid & 7) << 3;

  // issue one 64x64 bf16 tile pair (A,B) into buffer `buf`: 4 async ops/thread
#define GEMM_ISSUE(buf, k0)                                                    \
  do {                                                                         \
    _Pragma("unroll") for (int p = 0; p < 2; ++p) {                            \
      const int r = lrow + p * 32;                                             \
      async_copy_b128(lds_off32(&As[buf][r * LDT + lcol]),                     \
                      gA + (size_t)r * D_ + (k0) + lcol);                      \
      async_copy_b128(lds_off32(&Bs[buf][r * LDT + lcol]),                     \
                      gB + (size_t)r * D_ + (k0) + lcol);                      \
    }                                                                          \
  } while (0)

  GEMM_ISSUE(0, 0);
  for (int it = 0; it < D_ / 64; ++it) {
    const int cur = it & 1;
    if (it + 1 < D_ / 64) {
      GEMM_ISSUE(cur ^ 1, (it + 1) * 64);
      WAIT_ASYNC(0x4);   // drain current tile's 4 copies; next tile in flight
    } else {
      WAIT_ASYNC(0x0);
    }
    __syncthreads();
#pragma unroll
    for (int kk = 0; kk < 64; kk += 32) {
      v16bf a  = frag_a(&As[cur][wm * 16 * LDT + kk], LDT);
      v16bf b0 = frag_b(&Bs[cur][(wn * 32) * LDT + kk], LDT);
      v16bf b1 = frag_b(&Bs[cur][(wn * 32 + 16) * LDT + kk], LDT);
      acc0 = WMMA_BF16(a, b0, acc0);
      acc1 = WMMA_BF16(a, b1, acc1);
    }
    __syncthreads();
  }
#undef GEMM_ISSUE
}

// ---------------------------------------------------------------------------
// QKV projection: C[m][n] = sum_k X[m][k] * Wqkv[n][k], n in [0,2304)
// ---------------------------------------------------------------------------
__global__ __launch_bounds__(256) void qkv_gemm(
    const __bf16* __restrict__ X, const __bf16* __restrict__ W,
    const float* __restrict__ bq, const float* __restrict__ bk,
    const float* __restrict__ bv,
    __bf16* __restrict__ Qo, __bf16* __restrict__ Ko, __bf16* __restrict__ Vo) {
  __shared__ __bf16 As[2][64 * LDT];
  __shared__ __bf16 Bs[2][64 * LDT];
  const int tid = threadIdx.x;
  const int tn = blockIdx.x % 36;
  const int tm = blockIdx.x / 36;
  const int wave = tid >> 5, wm = wave >> 1, wn = wave & 1;
  const int lane = tid & 31, hi = (lane >> 4) & 1, cn = lane & 15;

  v8f acc0 = {}, acc1 = {};
  gemm64_async(X + (size_t)tm * 64 * D_, W + (size_t)tn * 64 * D_, As, Bs,
               acc0, acc1);

#pragma unroll
  for (int j = 0; j < 2; ++j) {
    const v8f acc = j ? acc1 : acc0;
    const int ng = tn * 64 + wn * 32 + j * 16 + cn;
    const int which = ng / D_;
    const int rem = ng % D_;
    const int h = rem >> 6, e = rem & 63;
    const float* bias = (which == 0) ? bq : (which == 1) ? bk : bv;
    __bf16* dst = (which == 0) ? Qo : (which == 1) ? Ko : Vo;
    const float bb = bias[rem];
#pragma unroll
    for (int r = 0; r < 8; ++r) {
      const int m = tm * 64 + wm * 16 + hi * 8 + r;
      const int b = m >> 10, s = m & 1023;
      dst[(((size_t)b * H_ + h) * S_ + s) * HD_ + e] = (__bf16)(acc[r] + bb);
    }
  }
}

// ---------------------------------------------------------------------------
// Flash attention: one block per (b, h, 64-row Q tile). Scores never hit HBM.
// ---------------------------------------------------------------------------
__global__ __launch_bounds__(256) void flash_attn(
    const __bf16* __restrict__ Q, const __bf16* __restrict__ K,
    const __bf16* __restrict__ V, __bf16* __restrict__ O) {
  __shared__ __bf16 Qs[64 * LDT];
  __shared__ __bf16 Ks[64 * LDT];
  __shared__ __bf16 Vt[64 * LDT];   // transposed: [e][t]
  __shared__ __bf16 Ps[64 * LDT];   // softmax probs bf16 [m][t]
  __shared__ float  Ss[64 * 68];    // raw scores f32
  __shared__ float  rowmax[64], rowsum[64], alpha_s[64], red[64 * 4];

  const int tid = threadIdx.x;
  const int qt = blockIdx.x & 15;
  const int h  = (blockIdx.x >> 4) % H_;
  const int b  = blockIdx.x / (16 * H_);
  const size_t head = ((size_t)b * H_ + h) * S_;
  const __bf16* gQ = Q + (head + (size_t)qt * 64) * HD_;

  const int wave = tid >> 5, wm = wave >> 1, wn = wave & 1;
  const int lane = tid & 31, hi = (lane >> 4) & 1, cn = lane & 15;
  const int lrow = tid >> 3, lcol = (tid & 7) << 3;

#pragma unroll
  for (int p = 0; p < 2; ++p) {
    const int r = lrow + p * 32;
    *(float4*)&Qs[r * LDT + lcol] = *(const float4*)(gQ + (size_t)r * HD_ + lcol);
  }
  if (tid < 64) { rowmax[tid] = -1e30f; rowsum[tid] = 0.0f; }

  v8f o0 = {}, o1 = {};
  const float scl = 0.125f;  // 1/sqrt(64)

  for (int kt = 0; kt < S_ / 64; ++kt) {
    const __bf16* gK = K + (head + (size_t)kt * 64) * HD_;
    const __bf16* gV = V + (head + (size_t)kt * 64) * HD_;
#pragma unroll
    for (int p = 0; p < 2; ++p) {
      const int r = lrow + p * 32;
      *(float4*)&Ks[r * LDT + lcol] = *(const float4*)(gK + (size_t)r * HD_ + lcol);
      float4 dv = *(const float4*)(gV + (size_t)r * HD_ + lcol);
      __bf16 tmp[8] __attribute__((aligned(16)));
      *(float4*)tmp = dv;
#pragma unroll
      for (int i = 0; i < 8; ++i) Vt[(lcol + i) * LDT + r] = tmp[i];  // transpose
    }
    // prefetch next K/V tiles toward L2 while softmax/WMMA run
    if (kt + 1 < S_ / 64) {
      __builtin_prefetch(gK + (size_t)64 * HD_ + (size_t)lrow * HD_ + lcol, 0, 1);
      __builtin_prefetch(gV + (size_t)64 * HD_ + (size_t)lrow * HD_ + lcol, 0, 1);
    }
    __syncthreads();

    // S = scl * Q K^T   (wave tile: rows wm*16, cols wn*32 + j*16)
#pragma unroll
    for (int j = 0; j < 2; ++j) {
      v8f sc = {};
#pragma unroll
      for (int kk = 0; kk < 64; kk += 32) {
        v16bf a  = frag_a(&Qs[wm * 16 * LDT + kk], LDT);
        v16bf bb = frag_b(&Ks[(wn * 32 + j * 16) * LDT + kk], LDT);
        sc = WMMA_BF16(a, bb, sc);
      }
#pragma unroll
      for (int r = 0; r < 8; ++r)
        Ss[(wm * 16 + hi * 8 + r) * 68 + wn * 32 + j * 16 + cn] = sc[r] * scl;
    }
    __syncthreads();

    // online softmax: 4 threads per row, 16 cols each
    const int row = tid >> 2, qq = tid & 3;
    float lmax = -1e30f;
#pragma unroll
    for (int c = 0; c < 16; ++c) lmax = fmaxf(lmax, Ss[row * 68 + qq * 16 + c]);
    red[row * 4 + qq] = lmax;
    __syncthreads();
    if (qq == 0) {
      float tmax = fmaxf(fmaxf(red[row * 4], red[row * 4 + 1]),
                         fmaxf(red[row * 4 + 2], red[row * 4 + 3]));
      float mold = rowmax[row];
      float mnew = fmaxf(mold, tmax);
      alpha_s[row] = __expf(mold - mnew);
      rowmax[row] = mnew;
    }
    __syncthreads();
    const float mrow = rowmax[row];
    float psum = 0.0f;
#pragma unroll
    for (int c = 0; c < 16; ++c) {
      float ev = __expf(Ss[row * 68 + qq * 16 + c] - mrow);
      Ps[row * LDT + qq * 16 + c] = (__bf16)ev;
      psum += ev;
    }
    red[row * 4 + qq] = psum;
    __syncthreads();
    if (qq == 0)
      rowsum[row] = rowsum[row] * alpha_s[row] + red[row * 4] + red[row * 4 + 1] +
                    red[row * 4 + 2] + red[row * 4 + 3];
    __syncthreads();

    // rescale running O, then O += P * V
#pragma unroll
    for (int r = 0; r < 8; ++r) {
      const float f = alpha_s[wm * 16 + hi * 8 + r];
      o0[r] *= f;
      o1[r] *= f;
    }
#pragma unroll
    for (int kk = 0; kk < 64; kk += 32) {
      v16bf a  = frag_a(&Ps[wm * 16 * LDT + kk], LDT);
      v16bf b0 = frag_b(&Vt[(wn * 32) * LDT + kk], LDT);
      v16bf b1 = frag_b(&Vt[(wn * 32 + 16) * LDT + kk], LDT);
      o0 = WMMA_BF16(a, b0, o0);
      o1 = WMMA_BF16(a, b1, o1);
    }
    __syncthreads();
  }

  // finalize: O /= rowsum, write concat layout [B][S][H][HD]
#pragma unroll
  for (int r = 0; r < 8; ++r) {
    const int m = wm * 16 + hi * 8 + r;
    const float inv = 1.0f / rowsum[m];
    const int s = qt * 64 + m;
#pragma unroll
    for (int j = 0; j < 2; ++j) {
      const float val = (j ? o1[r] : o0[r]) * inv;
      const int e = wn * 32 + j * 16 + cn;
      O[(((size_t)b * S_ + s) * H_ + h) * HD_ + e] = (__bf16)val;
    }
  }
}

// ---------------------------------------------------------------------------
// Output projection: out[m][n] = sum_k A[m][k] * Wo[n][k] + bo[n]  (fp32 out)
// ---------------------------------------------------------------------------
__global__ __launch_bounds__(256) void out_gemm(
    const __bf16* __restrict__ A, const __bf16* __restrict__ W,
    const float* __restrict__ bo, float* __restrict__ Out) {
  __shared__ __bf16 As[2][64 * LDT];
  __shared__ __bf16 Bs[2][64 * LDT];
  const int tid = threadIdx.x;
  const int tn = blockIdx.x % 12;
  const int tm = blockIdx.x / 12;
  const int wave = tid >> 5, wm = wave >> 1, wn = wave & 1;
  const int lane = tid & 31, hi = (lane >> 4) & 1, cn = lane & 15;

  v8f acc0 = {}, acc1 = {};
  gemm64_async(A + (size_t)tm * 64 * D_, W + (size_t)tn * 64 * D_, As, Bs,
               acc0, acc1);

#pragma unroll
  for (int j = 0; j < 2; ++j) {
    const v8f acc = j ? acc1 : acc0;
    const int ng = tn * 64 + wn * 32 + j * 16 + cn;
    const float bb = bo[ng];
#pragma unroll
    for (int r = 0; r < 8; ++r) {
      const int m = tm * 64 + wm * 16 + hi * 8 + r;
      Out[(size_t)m * D_ + ng] = acc[r] + bb;
    }
  }
}

// ---------------------------------------------------------------------------
extern "C" void kernel_launch(void* const* d_in, const int* in_sizes, int n_in,
                              void* d_out, int out_size, void* d_ws, size_t ws_size,
                              hipStream_t stream) {
  (void)in_sizes; (void)n_in; (void)out_size; (void)ws_size;
  const float* x  = (const float*)d_in[0];
  const float* Wq = (const float*)d_in[1];
  const float* bq = (const float*)d_in[2];
  const float* Wk = (const float*)d_in[3];
  const float* bk = (const float*)d_in[4];
  const float* Wv = (const float*)d_in[5];
  const float* bv = (const float*)d_in[6];
  const float* Wo = (const float*)d_in[7];
  const float* bo = (const float*)d_in[8];
  float* out = (float*)d_out;

  char* ws = (char*)d_ws;
  size_t off = 0;
  auto alloc = [&](size_t bytes) -> void* {
    void* p = ws + off;
    off = (off + bytes + 255) & ~(size_t)255;
    return p;
  };
  __bf16* xh   = (__bf16*)alloc((size_t)MTOT * D_ * 2);
  __bf16* Wqkv = (__bf16*)alloc((size_t)3 * D_ * D_ * 2);
  __bf16* Wob  = (__bf16*)alloc((size_t)D_ * D_ * 2);
  __bf16* Qb   = (__bf16*)alloc((size_t)MTOT * D_ * 2);
  __bf16* Kb   = (__bf16*)alloc((size_t)MTOT * D_ * 2);
  __bf16* Vb   = (__bf16*)alloc((size_t)MTOT * D_ * 2);
  __bf16* Ab   = (__bf16*)alloc((size_t)MTOT * D_ * 2);

  const int nx = MTOT * D_;      // 12,582,912
  const int nw = D_ * D_;        // 589,824
  cvt_bf16<<<(nx + 255) / 256, 256, 0, stream>>>(x, xh, nx);
  cvt_bf16<<<(nw + 255) / 256, 256, 0, stream>>>(Wq, Wqkv, nw);
  cvt_bf16<<<(nw + 255) / 256, 256, 0, stream>>>(Wk, Wqkv + (size_t)nw, nw);
  cvt_bf16<<<(nw + 255) / 256, 256, 0, stream>>>(Wv, Wqkv + (size_t)2 * nw, nw);
  cvt_bf16<<<(nw + 255) / 256, 256, 0, stream>>>(Wo, Wob, nw);

  qkv_gemm<<<(MTOT / 64) * (3 * D_ / 64), 256, 0, stream>>>(xh, Wqkv, bq, bk, bv,
                                                            Qb, Kb, Vb);
  flash_attn<<<B_ * H_ * (S_ / 64), 256, 0, stream>>>(Qb, Kb, Vb, Ab);
  out_gemm<<<(MTOT / 64) * (D_ / 64), 256, 0, stream>>>(Ab, Wob, bo, out);
}
